// DRew_GCN_63273458205289
// MI455X (gfx1250) — matrix operand order repaired and checked by reading, not compiled, not run on hardware
//
#include <hip/hip_runtime.h>

// ---------------- problem constants ----------------
#define N_      50000
#define IN_DIM  32
#define HID     128
#define OUT_DIM 10
#define NLAY    4
#define K5      5
#define EK_     400000
#define G_      500
#define NCONV   15
#define NH      (N_ * HID)          // 6,400,000 (divisible by 256)

typedef _Float16 half_t;
typedef __attribute__((ext_vector_type(16))) _Float16 v16h;
typedef __attribute__((ext_vector_type(8)))  _Float16 v8h;
typedef __attribute__((ext_vector_type(8)))  float    v8f;

// ---------------- utility kernels ----------------
__global__ void fill_kernel(float* __restrict__ p, float val, int total) {
    int i = blockIdx.x * blockDim.x + threadIdx.x;
    if (i < total) p[i] = val;
}

__global__ void cvt_w_kernel(const float* __restrict__ w, half_t* __restrict__ w16, int total) {
    int i = blockIdx.x * blockDim.x + threadIdx.x;
    if (i < total) w16[i] = (half_t)w[i];
}

// ---------------- embedding: h0 = x @ embW.T + embB ----------------
__global__ __launch_bounds__(HID)
void embed_kernel(const float* __restrict__ x, const float* __restrict__ Wt,
                  const float* __restrict__ b, float* __restrict__ h0,
                  half_t* __restrict__ h016) {
    __shared__ float sx[IN_DIM];
    const int n = blockIdx.x;
    const int o = threadIdx.x;
    if (o < IN_DIM) sx[o] = x[n * IN_DIM + o];
    __syncthreads();
    float acc = b[o];
    const float* wr = Wt + o * IN_DIM;
#pragma unroll
    for (int i = 0; i < IN_DIM; ++i) acc += sx[i] * wr[i];
    h0[n * HID + o]   = acc;
    h016[n * HID + o] = (half_t)acc;
}

// ---------------- degree / dinv (per edge-set k) ----------------
__global__ void deg_count_kernel(const int* __restrict__ dstAll, float* __restrict__ deg, int total) {
    int e = blockIdx.x * blockDim.x + threadIdx.x;
    if (e >= total) return;
    int k = e / EK_;
    atomicAdd(&deg[(size_t)k * N_ + dstAll[e]], 1.0f);
}

__global__ void deg_rsqrt_kernel(float* __restrict__ d, int total) {
    int i = blockIdx.x * blockDim.x + threadIdx.x;
    if (i < total) d[i] = rsqrtf(d[i]);
}

// ---------------- WMMA GEMM with fused self-loop/bias epilogue ----------------
// Hw = A(16-row strip,f16) @ W.T(f16) -> f32   (kept for the edge-scatter pass)
// acc (=|+=) (dinv^2 * Hw + bias) * invk       (fused; saves a full Hw re-read pass)
// Block = 256 threads = 8 waves; wave w computes one 16x16 tile at columns [16w,16w+16).
__global__ __launch_bounds__(256)
void conv_gemm_wmma(const half_t* __restrict__ A16, const half_t* __restrict__ W16,
                    const float* __restrict__ dinvk, const float* __restrict__ bias,
                    float* __restrict__ Hw, float* __restrict__ acc,
                    float invk, int first) {
    __shared__ __align__(32) half_t sA[16 * HID];     // 4 KB
    __shared__ __align__(32) half_t sB[HID * HID];    // 32 KB  (W[o][i] row-major)
    const int tid  = threadIdx.x;
    const int lane = tid & 31;
    const int wave = tid >> 5;
    const size_t rowBase = (size_t)blockIdx.x * 16;

    // stage A tile: 2048 halfs, 8 per thread (16B vector copies)
    {
        const v8h* g = (const v8h*)(A16 + rowBase * HID);
        ((v8h*)sA)[tid] = g[tid];
    }
    // stage full weight matrix: 16384 halfs, 64 per thread
    {
        const v8h* g = (const v8h*)W16;
        v8h* s = (v8h*)sB;
#pragma unroll
        for (int i = 0; i < 8; ++i) s[tid + 256 * i] = g[tid + 256 * i];
    }
    __syncthreads();

    const int mrow = lane & 15;     // A row within tile / B column within tile
    const int hi   = lane >> 4;     // half-wave selector
    v8f c = {};

#pragma unroll
    for (int t = 0; t < 4; ++t) {
        const int kb = t * 32;
        // A fragment (16x32 f16): lanes<16 hold K=0..15, lanes>=16 hold K=16..31
        v16h a = *(const v16h*)&sA[mrow * HID + kb + 16 * hi];
        // B fragment (32x16 f16): lane's column o, lanes<16 hold K=0..15, lanes>=16 K=16..31
        const int o = wave * 16 + mrow;
        v16h b = *(const v16h*)&sB[o * HID + kb + 16 * hi];
        c = __builtin_amdgcn_wmma_f32_16x16x32_f16(false, a, false, b, (short)0, c, false, false);
    }

    // C/D layout: VGPR r -> row (rowBase + r + 8*hi), col o = wave*16 + mrow
    const int o = wave * 16 + mrow;
    const float bo = bias[o] * invk;
    float* hrow = Hw  + (rowBase + (size_t)(8 * hi)) * HID + o;
    float* arow = acc + (rowBase + (size_t)(8 * hi)) * HID + o;
    const float* drow = dinvk + rowBase + (size_t)(8 * hi);
#pragma unroll
    for (int r = 0; r < 8; ++r) {
        float hv = c[r];
        hrow[(size_t)r * HID] = hv;
        float di = drow[r];
        float v  = di * di * invk * hv + bo;
        if (first) arow[(size_t)r * HID] = v;
        else       arow[(size_t)r * HID] += v;
    }
}

// ---------------- edge scatter: acc[dst] += Hw[src] * dinv[src]*dinv[dst] / k ----------------
// one thread per (edge, 4-feature chunk): float4 gather + 4 global f32 atomics
__global__ __launch_bounds__(256)
void edge_scatter_kernel(const float* __restrict__ Hw, const int* __restrict__ src,
                         const int* __restrict__ dst, const float* __restrict__ dinvk,
                         float* __restrict__ acc, float invk) {
    int gid = blockIdx.x * blockDim.x + threadIdx.x;
    int e = gid >> 5;
    if (e >= EK_) return;
    int chunk = gid & 31;
    int s = src[e], d = dst[e];
    float coef = dinvk[s] * dinvk[d] * invk;
    const float4 v = *(const float4*)(Hw + (size_t)s * HID + chunk * 4);
    float* ap = acc + (size_t)d * HID + chunk * 4;
    atomicAdd(ap + 0, v.x * coef);
    atomicAdd(ap + 1, v.y * coef);
    atomicAdd(ap + 2, v.z * coef);
    atomicAdd(ap + 3, v.w * coef);
}

// ---------------- ReLU + f16 shadow copy ----------------
__global__ void relu_kernel(const float* __restrict__ acc, float* __restrict__ h,
                            half_t* __restrict__ h16, int total) {
    int idx = blockIdx.x * blockDim.x + threadIdx.x;
    if (idx >= total) return;
    float v = fmaxf(acc[idx], 0.0f);
    h[idx] = v;
    h16[idx] = (half_t)v;
}

// ---------------- segment pooling (sum / max / count) ----------------
__global__ void pool_kernel(const float* __restrict__ h, const int* __restrict__ batch,
                            float* __restrict__ ssum, unsigned* __restrict__ smax,
                            float* __restrict__ cnt, int total) {
    int idx = blockIdx.x * blockDim.x + threadIdx.x;
    if (idx >= total) return;
    int n = idx >> 7, f = idx & (HID - 1);
    int g = batch[n];
    float v = h[idx];
    atomicAdd(&ssum[g * HID + f], v);
    // post-ReLU v >= 0, so uint bit-pattern ordering == float ordering; init 0 == 0.0f
    atomicMax(&smax[g * HID + f], __float_as_uint(v));
    if (f == 0) atomicAdd(&cnt[g], 1.0f);
}

// ---------------- MLP head ----------------
__global__ void head1_kernel(const float* __restrict__ ssum, const float* __restrict__ smaxf,
                             const float* __restrict__ cnt, const float* __restrict__ r1W,
                             const float* __restrict__ r1b, float* __restrict__ hid) {
    int idx = blockIdx.x * blockDim.x + threadIdx.x;
    const int HH = HID * 3 / 2;  // 192
    if (idx >= G_ * HH) return;
    int g = idx / HH, j = idx % HH;
    float inv = 1.0f / fmaxf(cnt[g], 1.0f);
    float a = r1b[j];
    const float* wr = r1W + (size_t)j * (3 * HID);
    const float* ss = ssum  + (size_t)g * HID;
    const float* sm = smaxf + (size_t)g * HID;
#pragma unroll 4
    for (int i = 0; i < HID; ++i) {
        a += wr[i] * ss[i];
        a += wr[HID + i] * sm[i];
        a += wr[2 * HID + i] * (ss[i] * inv);
    }
    hid[idx] = a > 0.0f ? a : 0.01f * a;   // leaky_relu(0.01)
}

__global__ void head2_kernel(const float* __restrict__ hid, const float* __restrict__ r2W,
                             const float* __restrict__ r2b, float* __restrict__ out) {
    int idx = blockIdx.x * blockDim.x + threadIdx.x;
    if (idx >= G_ * OUT_DIM) return;
    int g = idx / OUT_DIM, j = idx % OUT_DIM;
    const int HH = HID * 3 / 2;
    float a = r2b[j];
    const float* wr = r2W + (size_t)j * HH;
    const float* hr = hid + (size_t)g * HH;
#pragma unroll 4
    for (int i = 0; i < HH; ++i) a += wr[i] * hr[i];
    out[idx] = a;
}

// ---------------- host side ----------------
static inline char* wsalloc(char*& p, size_t bytes) {
    char* r = p;
    p += (bytes + 255) & ~(size_t)255;
    return r;
}

extern "C" void kernel_launch(void* const* d_in, const int* in_sizes, int n_in,
                              void* d_out, int out_size, void* d_ws, size_t ws_size,
                              hipStream_t stream) {
    (void)in_sizes; (void)n_in; (void)out_size; (void)ws_size;
    const float* x     = (const float*)d_in[0];
    const int*   kei   = (const int*)  d_in[1];   // (2, 5*EK) row-major
    /* d_in[2] = k_idx (unused) */
    const int*   batch = (const int*)  d_in[3];
    const float* embW  = (const float*)d_in[4];
    const float* embB  = (const float*)d_in[5];
    const float* convW = (const float*)d_in[6];
    const float* convB = (const float*)d_in[7];
    const float* r1W   = (const float*)d_in[8];
    const float* r1b   = (const float*)d_in[9];
    const float* r2W   = (const float*)d_in[10];
    const float* r2b   = (const float*)d_in[11];
    float* out = (float*)d_out;

    // workspace carve-out
    char* p = (char*)d_ws;
    float*  xl[NLAY + 2];
    half_t* xl16[NLAY + 2];
    for (int i = 0; i < NLAY + 2; ++i) xl[i]   = (float*) wsalloc(p, (size_t)NH * 4);
    for (int i = 0; i < NLAY + 2; ++i) xl16[i] = (half_t*)wsalloc(p, (size_t)NH * 2);
    float*  Hw   = (float*) wsalloc(p, (size_t)NH * 4);
    float*  acc  = (float*) wsalloc(p, (size_t)NH * 4);
    float*  dinv = (float*) wsalloc(p, (size_t)K5 * N_ * 4);
    half_t* W16  = (half_t*)wsalloc(p, (size_t)NCONV * HID * HID * 2);
    // pooling buffers contiguous -> single zero fill
    const int POOLF = G_ * (2 * HID + 1);            // ssum | smax | cnt
    float*  poolb = (float*) wsalloc(p, (size_t)POOLF * 4);
    float*  ssum = poolb;
    float*  smax = poolb + (size_t)G_ * HID;
    float*  cnt  = poolb + (size_t)2 * G_ * HID;
    float*  hid  = (float*) wsalloc(p, (size_t)G_ * (HID * 3 / 2) * 4);

    const int T = 256;
    const int* srcAll = kei;                       // row 0
    const int* dstAll = kei + (size_t)K5 * EK_;    // row 1

    // 1) embedding + f16 shadow
    embed_kernel<<<N_, HID, 0, stream>>>(x, embW, embB, xl[0], xl16[0]);

    // 2) f16 weights
    {
        int tot = NCONV * HID * HID;
        cvt_w_kernel<<<(tot + T - 1) / T, T, 0, stream>>>(convW, W16, tot);
    }

    // 3) per-k degrees -> dinv  (edge slices are layer-independent)
    fill_kernel<<<(K5 * N_ + T - 1) / T, T, 0, stream>>>(dinv, 1.0f, K5 * N_);
    deg_count_kernel<<<(K5 * EK_ + T - 1) / T, T, 0, stream>>>(dstAll, dinv, K5 * EK_);
    deg_rsqrt_kernel<<<(K5 * N_ + T - 1) / T, T, 0, stream>>>(dinv, K5 * N_);

    // 4) layers
    int ci = 0;
    for (int l = 0; l <= NLAY; ++l) {
        for (int k = 1; k <= l + 1; ++k, ++ci) {
            const half_t* inp16 = xl16[l + 1 - k];          // x_l[-k]
            const half_t* Wc    = W16 + (size_t)ci * HID * HID;
            const float*  dk    = dinv + (size_t)(k - 1) * N_;
            const float   invk  = 1.0f / (float)k;

            conv_gemm_wmma<<<N_ / 16, T, 0, stream>>>(
                inp16, Wc, dk, convB + (size_t)ci * HID, Hw, acc, invk, (k == 1) ? 1 : 0);
            edge_scatter_kernel<<<(EK_ * 32) / T, T, 0, stream>>>(
                Hw, srcAll + (size_t)(k - 1) * EK_, dstAll + (size_t)(k - 1) * EK_,
                dk, acc, invk);
        }
        relu_kernel<<<NH / T, T, 0, stream>>>(acc, xl[l + 1], xl16[l + 1], NH);
    }

    // 5) pooling
    fill_kernel<<<(POOLF + T - 1) / T, T, 0, stream>>>(poolb, 0.0f, POOLF);
    pool_kernel<<<NH / T, T, 0, stream>>>(xl[NLAY + 1], batch, ssum, (unsigned*)smax, cnt, NH);

    // 6) head
    head1_kernel<<<(G_ * 192 + T - 1) / T, T, 0, stream>>>(ssum, smax, cnt, r1W, r1b, hid);
    head2_kernel<<<(G_ * OUT_DIM + T - 1) / T, T, 0, stream>>>(hid, r2W, r2b, out);
}
